// Kernel_5453566362035708170_53472342835849
// MI455X (gfx1250) — compile-verified, matrix operand order checked by reading, not run on hardware
//
#include <hip/hip_runtime.h>

// ---------------------------------------------------------------------------
// CDNA5 (gfx1250) WMMA pipeline.
// Dominant GEMM: t11[n,c,c2*21+ks] = (1/64) * sum_hw t1[n,c,hw]*shift_ks(x)[n,c2,hw]
// Second big GEMM folded:  t13 = (t11 . W_g) . t2  = A . t2
// ---------------------------------------------------------------------------

#define N_B    8
#define C_CH   128
#define H_IMG  64
#define W_IMG  64
#define HW     4096          // 64*64
#define PH     66            // H + 2*1
#define PWP    72            // padded row stride for shifted copies (16B aligned)
#define T7W    76            // W + 2*6 (padded t7 row)
#define M21    21
#define CM     (C_CH * M21)  // 2688

typedef __bf16 bf16_t;
typedef __attribute__((ext_vector_type(16))) __bf16 v16bf;
typedef __attribute__((ext_vector_type(8)))  __bf16 v8bf;
typedef __attribute__((ext_vector_type(8)))  float  v8f;

static __device__ __forceinline__ unsigned short f32_to_bf16_raw(float f) {
  unsigned int u = __float_as_uint(f);
  unsigned int lsb = (u >> 16) & 1u;
  u += 0x7fffu + lsb;                 // round to nearest even
  return (unsigned short)(u >> 16);
}

// Load one 16(rows)x32(K) WMMA operand fragment from a row-major bf16 source.
// ISA 7.12.2 16-bit A layout: lane r=lane%16 holds row r; lanes<16 hold
// K={0..7,16..23}, lanes>=16 hold K={8..15,24..31}.  B operands loaded with the
// same routine from B^T ([N][K] row-major) storage.
static __device__ __forceinline__ v16bf load_frag(const unsigned short* src, int ld, int lane) {
  const int half = (lane >> 4) & 1;
  const int r = lane & 15;
  const unsigned short* row = src + (long)r * ld;
  v8bf lo = *(const v8bf*)(const void*)(row + half * 8);
  v8bf hi = *(const v8bf*)(const void*)(row + 16 + half * 8);
  v16bf out;
#pragma unroll
  for (int e = 0; e < 8; ++e) { out[e] = lo[e]; out[e + 8] = hi[e]; }
  return out;
}

// ---------------------------------------------------------------------------
// 1a) prep: t1 = p1*x (bf16), transposed bf16 relu(x) and t2 = x + t1
// ---------------------------------------------------------------------------
__global__ void k_prep(const float* __restrict__ x, const float* __restrict__ p1w,
                       unsigned short* __restrict__ t1bf,
                       unsigned short* __restrict__ xreluT, unsigned short* __restrict__ t2T) {
  long idx = (long)blockIdx.x * blockDim.x + threadIdx.x;
  if (idx >= (long)N_B * C_CH * HW) return;
  int hw = (int)(idx & (HW - 1));
  int cc = (int)((idx >> 12) & (C_CH - 1));
  int nn = (int)(idx >> 19);
  float v = x[idx];
  int y = hw >> 6;
  float p1 = p1w[cc * H_IMG + y];     // p1_w shape (1,c,h,1)
  float t1 = p1 * v;
  t1bf[idx] = f32_to_bf16_raw(t1);
  long tpos = ((long)nn * HW + hw) * C_CH + cc;
  xreluT[tpos] = f32_to_bf16_raw(v > 0.f ? v : 0.f);
  t2T[tpos]    = f32_to_bf16_raw(v + t1);
}

// ---------------------------------------------------------------------------
// 1b) 7 width-pre-shifted zero-padded bf16 copies of x: xpad7[sj][n][c][66][72]
//     Baking sj into the copy makes every GEMM staging load 16B-aligned.
// ---------------------------------------------------------------------------
__global__ void k_pad7(const float* __restrict__ x, unsigned short* __restrict__ xpad7) {
  long idx = (long)blockIdx.x * blockDim.x + threadIdx.x;
  const long total = 7L * N_B * C_CH * PH * PWP;
  if (idx >= total) return;
  int px = (int)(idx % PWP); long t = idx / PWP;
  int py = (int)(t % PH); t /= PH;
  int cc = (int)(t % C_CH); t /= C_CH;
  int nn = (int)(t % N_B); int sj = (int)(t / N_B);
  int y = py - 1, xx = px - 3 + sj;
  float v = 0.f;
  if (y >= 0 && y < H_IMG && xx >= 0 && xx < W_IMG)
    v = x[(long)(nn * C_CH + cc) * HW + y * W_IMG + xx];
  xpad7[idx] = f32_to_bf16_raw(v);
}

__global__ void k_cvt_p7(const float* __restrict__ p7, unsigned short* __restrict__ p7bf) {
  int i = blockIdx.x * blockDim.x + threadIdx.x;
  if (i < C_CH * C_CH) p7bf[i] = f32_to_bf16_raw(p7[i]);
}

__global__ void k_zero_t7(float* __restrict__ t7pad) {
  long i = (long)blockIdx.x * blockDim.x + threadIdx.x;
  if (i < (long)N_B * C_CH * H_IMG * T7W) t7pad[i] = 0.f;
}

// ---------------------------------------------------------------------------
// 2) t7[o,hw] = sum_c p7[o,c]*relu(x)[c,hw]   (WMMA, B staged in LDS)
// ---------------------------------------------------------------------------
__global__ void __launch_bounds__(256) k_t7(const unsigned short* __restrict__ p7bf,
                                            const unsigned short* __restrict__ xreluT,
                                            float* __restrict__ t7pad) {
  __shared__ __align__(16) unsigned short Bs[128 * 128]; // [N=hw][K=c]
  const int nn = blockIdx.y;
  const int hw0 = blockIdx.x * 128;
  const int tid = threadIdx.x;
  {
    const uint4* g = (const uint4*)(const void*)(xreluT + ((long)nn * HW + hw0) * C_CH);
    uint4* s = (uint4*)(void*)Bs;
#pragma unroll
    for (int e = 0; e < 8; ++e) s[tid + 256 * e] = g[tid + 256 * e];
  }
  __syncthreads();
  const int wave = tid >> 5, lane = tid & 31;
  v8f zero = {};
  v8f acc[8];
#pragma unroll
  for (int nt = 0; nt < 8; ++nt) acc[nt] = zero;
#pragma unroll
  for (int kc = 0; kc < 4; ++kc) {
    const int k0 = kc * 32;
    v16bf a = load_frag(p7bf + (long)(wave * 16) * C_CH + k0, C_CH, lane);
    v16bf bfr[8];
#pragma unroll
    for (int nt = 0; nt < 8; ++nt) bfr[nt] = load_frag(Bs + (nt * 16) * 128 + k0, 128, lane);
#pragma unroll
    for (int nt = 0; nt < 8; ++nt)
      acc[nt] = __builtin_amdgcn_wmma_f32_16x16x32_bf16(false, a, false, bfr[nt],
                                                        (short)0, acc[nt], false, false);
  }
  const int mB = (lane >> 4) * 8;   // C/D layout: M = vgpr + 8*(lane>=16)
  const int nIdx = lane & 15;
#pragma unroll
  for (int nt = 0; nt < 8; ++nt) {
    int hw = hw0 + nt * 16 + nIdx;
    int y = hw >> 6, xx = hw & 63;
#pragma unroll
    for (int r = 0; r < 8; ++r) {
      int o = wave * 16 + mB + r;
      t7pad[((long)(nn * C_CH + o) * H_IMG + y) * T7W + xx + 6] = acc[nt][r];
    }
  }
}

// ---------------------------------------------------------------------------
// 3) big GEMM with double-buffered LDS (K-chunk 64, 1 barrier/iter):
//    t11[c, c2*21+ks] = (1/64) sum_hw t1[c,hw] * x_shift_ks[c2,hw]
// ---------------------------------------------------------------------------
__global__ void __launch_bounds__(256) k_t11(const unsigned short* __restrict__ t1bf,
                                             const unsigned short* __restrict__ xpad7,
                                             float* __restrict__ t11) {
  __shared__ __align__(16) unsigned short Bs[2][128 * 64];  // [buf][N=c2][K=64]
  const int ks = blockIdx.x;               // shift 0..20
  const int nn = blockIdx.y;
  const int si = ks / 7, sj = ks % 7;      // row shift stays an offset; sj pre-baked
  const int tid = threadIdx.x;
  const int wave = tid >> 5, lane = tid & 31;
  const int rr = tid >> 1;                 // staging: row (c2)
  const int cb = (tid & 1) * 32;           // staging: K sub-chunk (32 elems = 4x uint4)
  // shifted, padded plane for (sj); rows 16B aligned (PWP=72 -> 144B stride)
  const unsigned short* xg =
      xpad7 + (((long)sj * N_B * C_CH + nn * C_CH + rr) * PH + si) * PWP;
  uint4* sd0 = (uint4*)(void*)(&Bs[0][rr * 64 + cb]);
  uint4* sd1 = (uint4*)(void*)(&Bs[1][rr * 64 + cb]);

  v8f zero = {};
  v8f acc[8];
#pragma unroll
  for (int nt = 0; nt < 8; ++nt) acc[nt] = zero;

  {   // prime chunk 0 (image row 0)
    const uint4* gs = (const uint4*)(const void*)(xg + cb);
#pragma unroll
    for (int e = 0; e < 4; ++e) sd0[e] = gs[e];
  }
  __syncthreads();

  for (int kc = 0; kc < 64; ++kc) {        // 4096 / 64 image rows
    const int cur = kc & 1;
    if (kc + 1 < 64) {                     // prefetch next row into other buffer
      const uint4* gs = (const uint4*)(const void*)(xg + (kc + 1) * PWP + cb);
      uint4* sd = cur ? sd0 : sd1;
#pragma unroll
      for (int e = 0; e < 4; ++e) sd[e] = gs[e];
    }
    const unsigned short* bp = Bs[cur];
#pragma unroll
    for (int kk = 0; kk < 2; ++kk) {
      const int k0 = kc * 64 + kk * 32;
      v16bf a = load_frag(t1bf + (long)(nn * C_CH + wave * 16) * HW + k0, HW, lane);
      v16bf bfr[8];
#pragma unroll
      for (int nt = 0; nt < 8; ++nt) bfr[nt] = load_frag(bp + (nt * 16) * 64 + kk * 32, 64, lane);
#pragma unroll
      for (int nt = 0; nt < 8; ++nt)
        acc[nt] = __builtin_amdgcn_wmma_f32_16x16x32_bf16(false, a, false, bfr[nt],
                                                          (short)0, acc[nt], false, false);
    }
    __syncthreads();
  }

  const int mB = (lane >> 4) * 8;
  const int nIdx = lane & 15;
#pragma unroll
  for (int nt = 0; nt < 8; ++nt) {
    int c2 = nt * 16 + nIdx;
#pragma unroll
    for (int r = 0; r < 8; ++r) {
      int c = wave * 16 + mB + r;
      t11[(long)(nn * C_CH + c) * CM + c2 * M21 + ks] = acc[nt][r] * (1.0f / 64.0f);
    }
  }
}

// ---------------------------------------------------------------------------
// 4) A[n,c,c'] = (1/sqrt(2688)) * sum_o t11[n,c,g*672+o] * p10[g*672+o, i]
// ---------------------------------------------------------------------------
__global__ void k_contract(const float* __restrict__ t11, const float* __restrict__ p10,
                           unsigned short* __restrict__ Abf) {
  int idx = blockIdx.x * blockDim.x + threadIdx.x;
  if (idx >= N_B * C_CH * C_CH) return;
  int cp = idx & 127; int t = idx >> 7;
  int c = t & 127; int nn = t >> 7;
  int g = cp >> 5, i = cp & 31;
  const float* trow = t11 + (long)(nn * C_CH + c) * CM + g * 672;
  const float* wrow = p10 + (long)g * 672 * 32 + i;
  float s = 0.f;
  for (int o = 0; o < 672; ++o) s += trow[o] * wrow[(long)o * 32];
  Abf[idx] = f32_to_bf16_raw(s * 0.019288065f);   // 1/sqrt(2688)
}

// ---------------------------------------------------------------------------
// 5) t13 = A . t2 (WMMA), fused epilogue: out = max(t13, x - mean5(t7 taps))
// ---------------------------------------------------------------------------
__global__ void __launch_bounds__(256) k_final(const unsigned short* __restrict__ Abf,
                                               const unsigned short* __restrict__ t2T,
                                               const float* __restrict__ x,
                                               const float* __restrict__ t7pad,
                                               float* __restrict__ out) {
  __shared__ __align__(16) unsigned short Bs[128 * 128]; // [N=hw][K=c']
  const int nn = blockIdx.y;
  const int hw0 = blockIdx.x * 128;
  const int tid = threadIdx.x;
  {
    const uint4* g = (const uint4*)(const void*)(t2T + ((long)nn * HW + hw0) * C_CH);
    uint4* s = (uint4*)(void*)Bs;
#pragma unroll
    for (int e = 0; e < 8; ++e) s[tid + 256 * e] = g[tid + 256 * e];
  }
  __syncthreads();
  const int wave = tid >> 5, lane = tid & 31;
  v8f zero = {};
  v8f acc[8];
#pragma unroll
  for (int nt = 0; nt < 8; ++nt) acc[nt] = zero;
#pragma unroll
  for (int kc = 0; kc < 4; ++kc) {
    const int k0 = kc * 32;
    v16bf a = load_frag(Abf + (long)(nn * C_CH + wave * 16) * C_CH + k0, C_CH, lane);
    v16bf bfr[8];
#pragma unroll
    for (int nt = 0; nt < 8; ++nt) bfr[nt] = load_frag(Bs + (nt * 16) * 128 + k0, 128, lane);
#pragma unroll
    for (int nt = 0; nt < 8; ++nt)
      acc[nt] = __builtin_amdgcn_wmma_f32_16x16x32_bf16(false, a, false, bfr[nt],
                                                        (short)0, acc[nt], false, false);
  }
  const int mB = (lane >> 4) * 8;
  const int nIdx = lane & 15;
#pragma unroll
  for (int nt = 0; nt < 8; ++nt) {
    int hw = hw0 + nt * 16 + nIdx;
    int y = hw >> 6, xx = hw & 63;
#pragma unroll
    for (int r = 0; r < 8; ++r) {
      int c = wave * 16 + mB + r;
      long b7 = ((long)(nn * C_CH + c) * H_IMG + y) * T7W + xx;  // tap j -> +3j
      float s5 = t7pad[b7] + t7pad[b7 + 3] + t7pad[b7 + 6] + t7pad[b7 + 9] + t7pad[b7 + 12];
      float xv = x[(long)(nn * C_CH + c) * HW + hw];
      float t12 = xv - 0.2f * s5;
      out[(long)(nn * C_CH + c) * HW + hw] = fmaxf(acc[nt][r], t12);
    }
  }
}

// ---------------------------------------------------------------------------
extern "C" void kernel_launch(void* const* d_in, const int* in_sizes, int n_in,
                              void* d_out, int out_size, void* d_ws, size_t ws_size,
                              hipStream_t stream) {
  (void)in_sizes; (void)n_in; (void)out_size; (void)ws_size;
  const float* x    = (const float*)d_in[0];
  const float* p1w  = (const float*)d_in[1];
  const float* p7w  = (const float*)d_in[2];
  const float* p10w = (const float*)d_in[3];
  float* out = (float*)d_out;

  char* p = (char*)d_ws;                               // all chunks 256B-multiples
  unsigned short* t1bf   = (unsigned short*)p; p += (size_t)N_B * C_CH * HW * 2;
  unsigned short* xpad7  = (unsigned short*)p; p += (size_t)7 * N_B * C_CH * PH * PWP * 2;
  unsigned short* xreluT = (unsigned short*)p; p += (size_t)N_B * HW * C_CH * 2;
  unsigned short* t2T    = (unsigned short*)p; p += (size_t)N_B * HW * C_CH * 2;
  unsigned short* p7bf   = (unsigned short*)p; p += (size_t)C_CH * C_CH * 2;
  float* t7pad = (float*)p; p += (size_t)N_B * C_CH * H_IMG * T7W * 4;
  float* t11   = (float*)p; p += (size_t)N_B * C_CH * CM * 4;
  unsigned short* Abf = (unsigned short*)p; p += (size_t)N_B * C_CH * C_CH * 2;

  long prepN = (long)N_B * C_CH * HW;
  k_prep<<<(int)((prepN + 255) / 256), 256, 0, stream>>>(x, p1w, t1bf, xreluT, t2T);
  long padN = 7L * N_B * C_CH * PH * PWP;
  k_pad7<<<(int)((padN + 255) / 256), 256, 0, stream>>>(x, xpad7);
  k_cvt_p7<<<(C_CH * C_CH + 255) / 256, 256, 0, stream>>>(p7w, p7bf);
  long zN = (long)N_B * C_CH * H_IMG * T7W;
  k_zero_t7<<<(int)((zN + 255) / 256), 256, 0, stream>>>(t7pad);
  k_t7<<<dim3(32, N_B), 256, 0, stream>>>(p7bf, xreluT, t7pad);
  k_t11<<<dim3(21, N_B), 256, 0, stream>>>(t1bf, xpad7, t11);
  k_contract<<<(N_B * C_CH * C_CH + 255) / 256, 256, 0, stream>>>(t11, p10w, Abf);
  k_final<<<dim3(32, N_B), 256, 0, stream>>>(Abf, t2T, x, t7pad, out);
}
